// RotatedNMSLayer_17995912970266
// MI455X (gfx1250) — compile-verified, matrix operand order checked by reading, not run on hardware
//
#include <hip/hip_runtime.h>

// Problem constants (fixed by reference setup): N = 4096+1024+256 = 5376
#define NBOX 5376
#define NW   168   // NBOX/32 : words per bitmask row, also K-tiles of 32
#define RT   336   // NBOX/16 : row tiles for WMMA rank kernel
#define IOU_T 0.25f

typedef __attribute__((ext_vector_type(16))) _Float16 v16h;
typedef __attribute__((ext_vector_type(8)))  float    v8f;

// ---------------- Kernel A: sigmoid / conf_m / box decode ----------------
__global__ void __launch_bounds__(256)
prep_kernel(const float* __restrict__ scores,
            const float* __restrict__ bboxes,
            const float* __restrict__ anchors,
            const float* __restrict__ strides,
            const float* __restrict__ thr,
            float* __restrict__ conf,
            float* __restrict__ confm,
            float* __restrict__ boxes) {
  int i = blockIdx.x * blockDim.x + threadIdx.x;
  if (i >= NBOX) return;
  float s = scores[i];
  float c = 1.0f / (1.0f + expf(-s));
  conf[i]  = c;
  confm[i] = (c >= thr[0]) ? c : -1.0f;
  float ax = anchors[2 * i], ay = anchors[2 * i + 1];
  float st = strides[i];
  float l = bboxes[4 * i + 0], t = bboxes[4 * i + 1];
  float r = bboxes[4 * i + 2], b = bboxes[4 * i + 3];
  boxes[4 * i + 0] = (ax - l) * st;
  boxes[4 * i + 1] = (ay - t) * st;
  boxes[4 * i + 2] = (ax + r) * st;
  boxes[4 * i + 3] = (ay + b) * st;
}

// ---------------- Kernel B: WMMA stable-sort ranks ----------------
// rank[i] = #{ j : cm[j] > cm[i]  ||  (cm[j] == cm[i] && j < i) }
// 16x32 f16 predicate tile (A) x all-ones B, accumulated in f32 C.
__global__ void __launch_bounds__(32)
rank_kernel(const float* __restrict__ confm, int* __restrict__ rank) {
  __shared__ float sc[NBOX];
  const int lane = threadIdx.x;
  for (int k = lane; k < NBOX; k += 32) sc[k] = confm[k];
  __syncthreads();

  const int M  = lane & 15;       // row within tile (A layout: lanes 0-15 / 16-31 both M=0..15)
  const int hi = lane >> 4;       // which K-halves this lane holds
  const int rowBase = blockIdx.x * 16;
  const int i  = rowBase + M;
  const float ci = sc[i];

  v16h bmat;
#pragma unroll
  for (int e = 0; e < 16; ++e) bmat[e] = (_Float16)1.0f;  // B = all ones -> D cols = rowsum(A)

  v8f acc = {};
  for (int kt = 0; kt < NW; ++kt) {
    const int kBase = kt * 32;
    v16h a;
#pragma unroll
    for (int e = 0; e < 16; ++e) {
      // 16-bit A 16x32 layout: lanes<16 hold K {0..7,16..23}; lanes>=16 hold K {8..15,24..31}
      int g = e >> 3, w = e & 7;
      int K = g * 16 + hi * 8 + w;
      int j = kBase + K;
      float cj = sc[j];
      bool pred = (cj > ci) || ((cj == ci) && (j < i));
      a[e] = pred ? (_Float16)1.0f : (_Float16)0.0f;
    }
    acc = __builtin_amdgcn_wmma_f32_16x16x32_f16(false, a, false, bmat,
                                                 (short)0, acc, false, false);
  }
  // C/D layout: VGPR r -> M=r (lanes 0-15), M=8+r (lanes 16-31); value uniform across N
  if (lane == 0) {
#pragma unroll
    for (int r = 0; r < 8; ++r) rank[rowBase + r] = (int)acc[r];
  } else if (lane == 16) {
#pragma unroll
    for (int r = 0; r < 8; ++r) rank[rowBase + 8 + r] = (int)acc[r];
  }
}

// ---------------- Kernel C: scatter into sorted order ----------------
__global__ void __launch_bounds__(256)
scatter_kernel(const int* __restrict__ rank,
               const float* __restrict__ boxes,
               const float* __restrict__ angles,
               const float* __restrict__ conf,
               const float* __restrict__ thr,
               float* __restrict__ sbox, float* __restrict__ sang,
               float* __restrict__ sconf, int* __restrict__ svalid) {
  int i = blockIdx.x * blockDim.x + threadIdx.x;
  if (i >= NBOX) return;
  int r = rank[i];
  sbox[4 * r + 0] = boxes[4 * i + 0];
  sbox[4 * r + 1] = boxes[4 * i + 1];
  sbox[4 * r + 2] = boxes[4 * i + 2];
  sbox[4 * r + 3] = boxes[4 * i + 3];
  sang[r]  = angles[i];
  float c  = conf[i];
  sconf[r] = c;
  svalid[r] = (c >= thr[0]) ? 1 : 0;
}

// ---------------- Kernel D: IoU suppression bitmask (wave32 ballot) ----------------
__global__ void __launch_bounds__(256)
mask_kernel(const float* __restrict__ sbox, unsigned int* __restrict__ mask) {
  const int s    = blockIdx.x;
  const int lane = threadIdx.x & 31;
  const int wv   = threadIdx.x >> 5;
  const float ax1 = sbox[4 * s + 0], ay1 = sbox[4 * s + 1];
  const float ax2 = sbox[4 * s + 2], ay2 = sbox[4 * s + 3];
  const float areaA = (ax2 - ax1) * (ay2 - ay1);
  for (int w = wv; w < NW; w += 8) {
    int t = w * 32 + lane;
    float bx1 = sbox[4 * t + 0], by1 = sbox[4 * t + 1];
    float bx2 = sbox[4 * t + 2], by2 = sbox[4 * t + 3];
    float areaB = (bx2 - bx1) * (by2 - by1);
    float iw = fminf(ax2, bx2) - fmaxf(ax1, bx1); iw = fmaxf(iw, 0.0f);
    float ih = fminf(ay2, by2) - fmaxf(ay1, by1); ih = fmaxf(ih, 0.0f);
    float inter = iw * ih;
    float iou = inter / (areaA + areaB - inter);
    int pred = (t < s) && (iou > IOU_T);           // only already-processed cols
    unsigned long long bal = __ballot(pred);       // wave32: low 32 bits
    if (lane == 0) mask[(size_t)s * NW + w] = (unsigned int)bal;
  }
}

// ---------------- Kernel E: serial greedy NMS, single wave ----------------
__global__ void __launch_bounds__(32)
nms_kernel(const unsigned int* __restrict__ mask,
           const int* __restrict__ svalid, int* __restrict__ keep) {
  __shared__ unsigned int kw[NW];
  volatile unsigned int* vk = kw;
  const int lane = threadIdx.x;
  for (int w = lane; w < NW; w += 32) kw[w] = 0u;
  __syncthreads();
  for (int s = 0; s < NBOX; ++s) {
    unsigned int acc = 0u;
    const unsigned int* row = mask + (size_t)s * NW;
    for (int w = lane; w < NW; w += 32) acc |= (vk[w] & row[w]);
    int supp = __any(acc != 0u);
    if (lane == 0) {
      int k = (svalid[s] != 0) && !supp;
      keep[s] = k;
      if (k) vk[s >> 5] |= (1u << (s & 31));
    }
    __syncthreads();
  }
}

// ---------------- Kernel F: emit (N,7) output ----------------
__global__ void __launch_bounds__(256)
out_kernel(const float* __restrict__ sbox, const float* __restrict__ sang,
           const float* __restrict__ sconf, const int* __restrict__ keep,
           float* __restrict__ out) {
  int s = blockIdx.x * blockDim.x + threadIdx.x;
  if (s >= NBOX) return;
  float o0 = 0.f, o1 = 0.f, o2 = 0.f, o3 = 0.f, o4 = 0.f, o5 = 0.f, o6 = 0.f;
  if (keep[s]) {
    o0 = sbox[4 * s + 0]; o1 = sbox[4 * s + 1];
    o2 = sbox[4 * s + 2]; o3 = sbox[4 * s + 3];
    o4 = sang[s]; o5 = sconf[s]; o6 = 0.0f;   // cls == 0 (single class)
  }
  out[7 * s + 0] = o0; out[7 * s + 1] = o1; out[7 * s + 2] = o2;
  out[7 * s + 3] = o3; out[7 * s + 4] = o4; out[7 * s + 5] = o5;
  out[7 * s + 6] = o6;
}

extern "C" void kernel_launch(void* const* d_in, const int* in_sizes, int n_in,
                              void* d_out, int out_size, void* d_ws, size_t ws_size,
                              hipStream_t stream) {
  (void)in_sizes; (void)n_in; (void)out_size; (void)ws_size;
  const float* scores  = (const float*)d_in[0];
  const float* bboxes  = (const float*)d_in[1];
  const float* angles  = (const float*)d_in[2];
  const float* anchors = (const float*)d_in[3];
  const float* strides = (const float*)d_in[4];
  const float* thr     = (const float*)d_in[5];
  float* out = (float*)d_out;

  // workspace carve-up (~3.9 MB total)
  char* ws = (char*)d_ws;
  size_t off = 0;
  auto carve = [&](size_t bytes) { char* p = ws + off; off += (bytes + 255) & ~(size_t)255; return p; };
  float*        conf   = (float*)carve(NBOX * 4);
  float*        confm  = (float*)carve(NBOX * 4);
  float*        boxes  = (float*)carve(NBOX * 16);
  int*          rank   = (int*)  carve(NBOX * 4);
  float*        sbox   = (float*)carve(NBOX * 16);
  float*        sang   = (float*)carve(NBOX * 4);
  float*        sconf  = (float*)carve(NBOX * 4);
  int*          svalid = (int*)  carve(NBOX * 4);
  int*          keep   = (int*)  carve(NBOX * 4);
  unsigned int* mask   = (unsigned int*)carve((size_t)NBOX * NW * 4);

  const int TB = 256;
  const int NB = (NBOX + TB - 1) / TB;   // 21

  prep_kernel<<<NB, TB, 0, stream>>>(scores, bboxes, anchors, strides, thr,
                                     conf, confm, boxes);
  rank_kernel<<<RT, 32, 0, stream>>>(confm, rank);
  scatter_kernel<<<NB, TB, 0, stream>>>(rank, boxes, angles, conf, thr,
                                        sbox, sang, sconf, svalid);
  mask_kernel<<<NBOX, TB, 0, stream>>>(sbox, mask);
  nms_kernel<<<1, 32, 0, stream>>>(mask, svalid, keep);
  out_kernel<<<NB, TB, 0, stream>>>(sbox, sang, sconf, keep, out);
}
